// Net_89687507075936
// MI455X (gfx1250) — matrix operand order repaired and checked by reading, not compiled, not run on hardware
//
#include <hip/hip_runtime.h>

typedef unsigned short u16;
typedef unsigned int   u32;

typedef __attribute__((ext_vector_type(16))) __bf16 v16bf;
typedef __attribute__((ext_vector_type(8)))  float  v8f;

union BF16x16 { v16bf v; u32 u[8]; };

// LDS row stride in bf16 elems: 80B rows -> 16B aligned (async B128 stores),
// and row*20 dwords mod 64 banks is distinct for 16 consecutive rows
// (conflict-free fragment reads).
#define ASTR 40

__device__ __forceinline__ u16 f2bf(float f){
  u32 u = __float_as_uint(f);
  u32 r = (u + 0x7FFFu + ((u >> 16) & 1u)) >> 16;  // RNE
  return (u16)r;
}
__device__ __forceinline__ float bf2f(u16 h){
  return __uint_as_float(((u32)h) << 16);
}
__device__ __forceinline__ float sigmf(float x){ return 1.0f/(1.0f + __expf(-x)); }

// Issue async global->LDS copies for one 128x32 bf16 A tile and 64x32 W tile.
// 128 threads; each thread moves 4x16B (A) + 2x16B (W) => 6 async instrs/wave.
__device__ __forceinline__ void issue_tile(const u16* __restrict__ A, int lda,
                                           const u16* __restrict__ W, int ldw,
                                           size_t mtile, size_t ntile, int k0,
                                           u16* Asb, u16* Wsb, int t)
{
  #pragma unroll
  for (int u = 0; u < 4; u++) {
    int uu  = t + u * 128;            // 0..511 transfer units of 16B
    int row = uu >> 2;
    int c0  = (uu & 3) * 8;           // bf16 elems
    const u16* ga = A + (mtile + row) * (size_t)lda + k0 + c0;
    u32 la = (u32)(uintptr_t)(Asb + row * ASTR + c0);
    asm volatile("global_load_async_to_lds_b128 %0, %1, off"
                 :: "v"(la), "v"(ga) : "memory");
  }
  #pragma unroll
  for (int u = 0; u < 2; u++) {
    int uu  = t + u * 128;            // 0..255 transfer units of 16B
    int row = uu >> 2;
    int c0  = (uu & 3) * 8;
    const u16* gw = W + (ntile + row) * (size_t)ldw + k0 + c0;
    u32 lw = (u32)(uintptr_t)(Wsb + row * ASTR + c0);
    asm volatile("global_load_async_to_lds_b128 %0, %1, off"
                 :: "v"(lw), "v"(gw) : "memory");
  }
}

// ---------------------------------------------------------------------------
// Generic 128x64 bf16 GEMM tile: C_tile = A[mtile:+128,:K] * W[ntile:+64,:K]^T
// A row-major [M][lda] bf16, W row-major [N][ldw] bf16 (i.e. B = W^T).
// Block = 128 threads = 4 waves; wave w computes rows [w*32, w*32+32)
// (two 16-row strips, so each B fragment feeds two WMMAs).
// Double-buffered LDS fed by GLOBAL_LOAD_ASYNC_TO_LDS_B128 (ASYNCcnt).
// acc[ms][ns]: M-strip ms (0..1), N-subtile ns (0..3).
// ---------------------------------------------------------------------------
__device__ __forceinline__ void gemm_tile(const u16* __restrict__ A, int lda,
                                          const u16* __restrict__ W, int ldw,
                                          int K, size_t mtile, size_t ntile,
                                          v8f acc[2][4])
{
  __shared__ __align__(16) u16 As[2][128 * ASTR];
  __shared__ __align__(16) u16 Ws[2][64 * ASTR];
  const int t    = threadIdx.x;
  const int wave = t >> 5, lane = t & 31, half = lane >> 4, l15 = lane & 15;

  v8f zero = {0.f,0.f,0.f,0.f,0.f,0.f,0.f,0.f};
  #pragma unroll
  for (int ms = 0; ms < 2; ms++)
    #pragma unroll
    for (int ns = 0; ns < 4; ns++) acc[ms][ns] = zero;

  const int nt = K >> 5;
  issue_tile(A, lda, W, ldw, mtile, ntile, 0, As[0], Ws[0], t);

  for (int i = 0; i < nt; i++) {
    const int buf = i & 1;
    if (i + 1 < nt) {
      // prefetch next K-tile into the other buffer while this one is consumed
      issue_tile(A, lda, W, ldw, mtile, ntile, (i + 1) << 5,
                 As[buf ^ 1], Ws[buf ^ 1], t);
      asm volatile("s_wait_asynccnt 0x6" ::: "memory");  // tile i landed
    } else {
      asm volatile("s_wait_asynccnt 0x0" ::: "memory");
    }
    __syncthreads();   // all waves' async writes for tile i visible

    // A fragments: lane holds row (wave*32 + ms*16 + l15); K pairs per ISA map
    BF16x16 a[2];
    #pragma unroll
    for (int ms = 0; ms < 2; ms++)
      #pragma unroll
      for (int j = 0; j < 8; j++) {
        int kb = ((j & 4) ? 16 : 0) + half * 8 + ((j & 3) << 1);
        a[ms].u[j] = *(const u32*)(&As[buf][(wave*32 + ms*16 + l15) * ASTR + kb]);
      }
    #pragma unroll
    for (int ns = 0; ns < 4; ns++) {
      BF16x16 b;
      #pragma unroll
      for (int j = 0; j < 8; j++)
        b.u[j] = *(const u32*)(&Ws[buf][(ns*16 + l15) * ASTR + ((j & 4) ? 16 : 0) + half*8 + ((j & 3) << 1)]);
      acc[0][ns] = __builtin_amdgcn_wmma_f32_16x16x32_bf16(
          false, a[0].v, false, b.v, (short)0, acc[0][ns], false, false);
      acc[1][ns] = __builtin_amdgcn_wmma_f32_16x16x32_bf16(
          false, a[1].v, false, b.v, (short)0, acc[1][ns], false, false);
    }
    __syncthreads();   // done reading buf before it is refilled next+1 iter
  }
}

// C = A * W^T (+bias), fp32 out
__global__ void k_gemm_plain(const u16* __restrict__ A, int lda,
                             const u16* __restrict__ W, int ldw, int K,
                             const float* __restrict__ bias,
                             float* __restrict__ C, int ldc)
{
  size_t mtile = (size_t)blockIdx.y * 128, ntile = (size_t)blockIdx.x * 64;
  v8f acc[2][4];
  gemm_tile(A, lda, W, ldw, K, mtile, ntile, acc);
  int t = threadIdx.x, wave = t >> 5, lane = t & 31, half = lane >> 4, l15 = lane & 15;
  #pragma unroll
  for (int ns = 0; ns < 4; ns++) {
    size_t col = ntile + ns*16 + l15;
    float bv = bias ? bias[col] : 0.f;
    #pragma unroll
    for (int ms = 0; ms < 2; ms++)
      #pragma unroll
      for (int r = 0; r < 8; r++) {
        size_t m = mtile + wave*32 + ms*16 + half*8 + r;
        C[m * (size_t)ldc + col] = acc[ms][ns][r] + bv;
      }
  }
}

// pointwise-conv pass 1: per-column sum/sumsq for BN2 (no C store)
__global__ void k_gemm_stats(const u16* __restrict__ A, int lda,
                             const u16* __restrict__ W, int ldw, int K,
                             float* __restrict__ stats, int N)
{
  size_t mtile = (size_t)blockIdx.y * 128, ntile = (size_t)blockIdx.x * 64;
  v8f acc[2][4];
  gemm_tile(A, lda, W, ldw, K, mtile, ntile, acc);
  int t = threadIdx.x, lane = t & 31, l15 = lane & 15;
  __shared__ float csum[64], csq[64];
  if (t < 64) { csum[t] = 0.f; csq[t] = 0.f; }
  __syncthreads();
  #pragma unroll
  for (int ns = 0; ns < 4; ns++) {
    float s = 0.f, q = 0.f;
    #pragma unroll
    for (int ms = 0; ms < 2; ms++)
      #pragma unroll
      for (int r = 0; r < 8; r++) { float v = acc[ms][ns][r]; s += v; q += v*v; }
    atomicAdd(&csum[ns*16 + l15], s);
    atomicAdd(&csq [ns*16 + l15], q);
  }
  __syncthreads();
  if (t < 64) {
    atomicAdd(&stats[ntile + t],     csum[t]);
    atomicAdd(&stats[N + ntile + t], csq[t]);
  }
}

// pointwise-conv pass 2: relu(bn2(C)) pooled over spatial (P=12544 % 128 == 0,
// so each 128-row tile lies in one batch image)
__global__ void k_gemm_pool(const u16* __restrict__ A,
                            const u16* __restrict__ W, int K,
                            const float* __restrict__ sc, const float* __restrict__ sh,
                            float* __restrict__ pool)
{
  size_t mtile = (size_t)blockIdx.y * 128, ntile = (size_t)blockIdx.x * 64;
  v8f acc[2][4];
  gemm_tile(A, 256, W, 256, K, mtile, ntile, acc);
  int t = threadIdx.x, lane = t & 31, l15 = lane & 15;
  __shared__ float csum[64];
  if (t < 64) csum[t] = 0.f;
  __syncthreads();
  int b = (int)(mtile / 12544);
  #pragma unroll
  for (int ns = 0; ns < 4; ns++) {
    size_t col = ntile + ns*16 + l15;
    float scv = sc[col], shv = sh[col], s = 0.f;
    #pragma unroll
    for (int ms = 0; ms < 2; ms++)
      #pragma unroll
      for (int r = 0; r < 8; r++) {
        float v = acc[ms][ns][r] * scv + shv;
        s += (v > 0.f) ? v : 0.f;
      }
    atomicAdd(&csum[ns*16 + l15], s);
  }
  __syncthreads();
  if (t < 64) atomicAdd(&pool[(size_t)b*512 + ntile + t], csum[t]);
}

// ---------------------------------------------------------------------------
// conv1 3->256, 3x3 s2 p1 on 224x224 -> 112x112, bf16 out + BN1 stats
// grid (49, 256, 16), block 256
// ---------------------------------------------------------------------------
__global__ void k_conv1(const float* __restrict__ img, const float* __restrict__ w,
                        u16* __restrict__ raw, float* __restrict__ stats)
{
  int b = blockIdx.z, c = blockIdx.y, t = threadIdx.x;
  int p = blockIdx.x * 256 + t;
  __shared__ float wsm[27];
  if (t < 27) wsm[t] = w[c*27 + t];
  __syncthreads();
  int oy = p / 112, ox = p % 112;
  float acc = 0.f;
  #pragma unroll
  for (int ic = 0; ic < 3; ic++)
    #pragma unroll
    for (int ky = 0; ky < 3; ky++) {
      int iy = oy*2 - 1 + ky;
      if (iy < 0 || iy > 223) continue;
      #pragma unroll
      for (int kx = 0; kx < 3; kx++) {
        int ix = ox*2 - 1 + kx;
        if (ix < 0 || ix > 223) continue;
        acc += img[(((size_t)b*3 + ic)*224 + iy)*224 + ix] * wsm[ic*9 + ky*3 + kx];
      }
    }
  raw[((size_t)b*256 + c)*12544 + p] = f2bf(acc);
  __shared__ float rs[256], rq[256];
  rs[t] = acc; rq[t] = acc*acc;
  __syncthreads();
  for (int s = 128; s > 0; s >>= 1) {
    if (t < s) { rs[t] += rs[t+s]; rq[t] += rq[t+s]; }
    __syncthreads();
  }
  if (t == 0) { atomicAdd(&stats[c], rs[0]); atomicAdd(&stats[256 + c], rq[0]); }
}

// finalize BN: scale = g*rsqrt(var+eps), shift = b - mean*scale
__global__ void k_bnfin(const float* __restrict__ stats, const float* __restrict__ g,
                        const float* __restrict__ bb, float* __restrict__ scale,
                        float* __restrict__ shift, int C, float invc)
{
  int c = blockIdx.x * blockDim.x + threadIdx.x;
  if (c >= C) return;
  float m = stats[c] * invc;
  float v = stats[C + c] * invc - m*m;
  float s = g[c] * rsqrtf(v + 1e-5f);
  scale[c] = s; shift[c] = bb[c] - m*s;
}

// depthwise 3x3 s1 p1 on relu(bn1(conv1)); writes GEMM-A layout [(b*P+p)][c]
__global__ void k_dw(const u16* __restrict__ raw, const float* __restrict__ dww,
                     const float* __restrict__ sc, const float* __restrict__ sh,
                     u16* __restrict__ dwA)
{
  int b = blockIdx.z, c = blockIdx.y, t = threadIdx.x;
  int p = blockIdx.x * 256 + t;
  __shared__ float w9[9], s2[2];
  if (t < 9) w9[t] = dww[c*9 + t];
  if (t == 9)  s2[0] = sc[c];
  if (t == 10) s2[1] = sh[c];
  __syncthreads();
  int oy = p / 112, ox = p % 112;
  const u16* base = raw + ((size_t)b*256 + c)*12544;
  float acc = 0.f;
  #pragma unroll
  for (int ky = 0; ky < 3; ky++) {
    int iy = oy - 1 + ky;
    if (iy < 0 || iy > 111) continue;
    #pragma unroll
    for (int kx = 0; kx < 3; kx++) {
      int ix = ox - 1 + kx;
      if (ix < 0 || ix > 111) continue;
      float v = bf2f(base[iy*112 + ix]) * s2[0] + s2[1];
      acc += ((v > 0.f) ? v : 0.f) * w9[ky*3 + kx];
    }
  }
  dwA[((size_t)b*12544 + p)*256 + c] = f2bf(acc);
}

__global__ void k_cvt(const float* __restrict__ s, u16* __restrict__ d, int n)
{
  int i = blockIdx.x * 256 + threadIdx.x;
  if (i < n) d[i] = f2bf(s[i]);
}
__global__ void k_zero(float* __restrict__ p, int n)
{
  int i = blockIdx.x * 256 + threadIdx.x;
  if (i < n) p[i] = 0.f;
}
__global__ void k_scale(float* __restrict__ p, float s, int n)
{
  int i = blockIdx.x * 256 + threadIdx.x;
  if (i < n) p[i] *= s;
}

// SE + enc_fc + ctx (all tiny, one block per batch row)
__global__ void k_se(const float* __restrict__ pooled, const float* __restrict__ w1,
                     const float* __restrict__ w2, const float* __restrict__ encw,
                     const float* __restrict__ encb, const float* __restrict__ vw,
                     float* __restrict__ ctx)
{
  int b = blockIdx.x, t = threadIdx.x;   // 128 threads
  __shared__ float pb[512], y1[128], f[512], fe[512];
  for (int i = t; i < 512; i += 128) pb[i] = pooled[(size_t)b*512 + i];
  __syncthreads();
  { float a = 0.f;
    for (int k = 0; k < 512; k++) a += pb[k] * w1[(size_t)t*512 + k];
    y1[t] = (a > 0.f) ? a : 0.f; }
  __syncthreads();
  for (int i = t; i < 512; i += 128) {
    float a = 0.f;
    for (int k = 0; k < 128; k++) a += y1[k] * w2[(size_t)i*128 + k];
    f[i] = pb[i] * sigmf(a);
  }
  __syncthreads();
  for (int i = t; i < 512; i += 128) {
    float a = encb[i];
    for (int k = 0; k < 512; k++) a += f[k] * encw[(size_t)i*512 + k];
    fe[i] = a;
  }
  __syncthreads();
  for (int i = t; i < 512; i += 128) {
    float a = 0.f;
    for (int k = 0; k < 512; k++) a += fe[k] * vw[(size_t)i*512 + k];
    ctx[(size_t)b*512 + i] = a;
  }
}

// gctx = ctx @ Wih[:,512:].T + b_ih (time-invariant part of GRU input gates)
__global__ void k_gctx(const float* __restrict__ ctx, const float* __restrict__ wih,
                       const float* __restrict__ bih, float* __restrict__ g)
{
  int b = blockIdx.y, j = blockIdx.x * 256 + threadIdx.x;  // 6*256 = 1536
  __shared__ float cs[512];
  for (int i = threadIdx.x; i < 512; i += 256) cs[i] = ctx[(size_t)b*512 + i];
  __syncthreads();
  float a = bih[j];
  for (int k = 0; k < 512; k++) a += cs[k] * wih[(size_t)j*1024 + 512 + k];
  g[(size_t)b*1536 + j] = a;
}

// gather embeddings (bf16) for batched WMMA input projection
__global__ void k_embed(const int* __restrict__ cap, const float* __restrict__ embed,
                        u16* __restrict__ aemb)
{
  int m = blockIdx.x;                 // 0..511 = b*32 + t
  int b = m >> 5, tt = m & 31;
  int tok = cap[b*33 + tt];
  for (int i = threadIdx.x; i < 512; i += 256)
    aemb[(size_t)m*512 + i] = f2bf(embed[(size_t)tok*512 + i]);
}

// one GRU step for all 16 batch rows; stores h and bf16 hall for logits GEMM
__global__ void k_gru(const float* __restrict__ gi, const float* __restrict__ gctx,
                      const float* __restrict__ whh, const float* __restrict__ bhh,
                      float* __restrict__ h, u16* __restrict__ hall, int tstep)
{
  int b = blockIdx.x, j = threadIdx.x;  // 512 threads
  __shared__ float hs[512];
  hs[j] = h[(size_t)b*512 + j];
  __syncthreads();
  float ar = bhh[j], az = bhh[512 + j], an = bhh[1024 + j];
  for (int k = 0; k < 512; k++) {
    float hv = hs[k];
    ar += hv * whh[(size_t)j*512 + k];
    az += hv * whh[(size_t)(512 + j)*512 + k];
    an += hv * whh[(size_t)(1024 + j)*512 + k];
  }
  int m = b*32 + tstep;
  const float* gie = gi + (size_t)m*1536;
  const float* gcx = gctx + (size_t)b*1536;
  float r = sigmf(gie[j]        + gcx[j]        + ar);
  float z = sigmf(gie[512 + j]  + gcx[512 + j]  + az);
  float n = tanhf(gie[1024 + j] + gcx[1024 + j] + r*an);
  float hn = (1.f - z)*n + z*hs[j];
  h[(size_t)b*512 + j] = hn;
  hall[(size_t)m*512 + j] = f2bf(hn);
}

// ---------------------------------------------------------------------------
extern "C" void kernel_launch(void* const* d_in, const int* in_sizes, int n_in,
                              void* d_out, int out_size, void* d_ws, size_t ws_size,
                              hipStream_t stream)
{
  const float* images  = (const float*)d_in[0];
  const int*   caps    = (const int*)  d_in[1];
  const float* conv1w  = (const float*)d_in[2];
  const float* bn1g    = (const float*)d_in[3];
  const float* bn1b    = (const float*)d_in[4];
  const float* dww     = (const float*)d_in[5];
  const float* pww     = (const float*)d_in[6];
  const float* bn2g    = (const float*)d_in[7];
  const float* bn2b    = (const float*)d_in[8];
  const float* se1     = (const float*)d_in[9];
  const float* se2     = (const float*)d_in[10];
  const float* encw    = (const float*)d_in[11];
  const float* encb    = (const float*)d_in[12];
  const float* embed   = (const float*)d_in[13];
  const float* vw      = (const float*)d_in[16];
  const float* wih     = (const float*)d_in[17];
  const float* whh     = (const float*)d_in[18];
  const float* bih     = (const float*)d_in[19];
  const float* bhh     = (const float*)d_in[20];
  const float* fcw     = (const float*)d_in[21];
  const float* fcb     = (const float*)d_in[22];
  float* out = (float*)d_out;

  char* W8 = (char*)d_ws;
  // region A (102,760,448 B): conv1 raw bf16, later reused for decoder buffers
  u16*  conv1raw = (u16*)(W8 + 0);
  u16*  dwA      = (u16*)(W8 + 102760448);      // 102,760,448 B
  char* SM       = W8 + 205520896;              // small persistent region
  float* stats1 = (float*)(SM);                 // 512 f
  float* sc1    = (float*)(SM + 2048);
  float* sh1    = (float*)(SM + 3072);
  float* stats2 = (float*)(SM + 4096);          // 1024 f
  float* sc2    = (float*)(SM + 8192);
  float* sh2    = (float*)(SM + 10240);
  float* pool   = (float*)(SM + 12288);         // 8192 f
  float* ctx    = (float*)(SM + 45056);         // 8192 f
  u16*   pwW    = (u16*)  (SM + 77824);         // 131072 bf16
  // region A phase 2 (valid after k_dw finished reading conv1raw)
  u16*   fcW    = (u16*)  (W8 + 0);             // 16,384,000 bf16
  u16*   wihB   = (u16*)  (W8 + 32768000);      // 1,572,864 bf16
  u16*   aemb   = (u16*)  (W8 + 35913728);      // 512*512 bf16
  float* giemb  = (float*)(W8 + 36438016);      // 512*1536 f
  u16*   hall   = (u16*)  (W8 + 39583744);      // 512*512 bf16
  float* hbuf   = (float*)(W8 + 40108032);      // 16*512 f
  float* gctx   = (float*)(W8 + 40140800);      // 16*1536 f

  const float invCnt = 1.f / 200704.f;   // B * 112*112

  // --- encoder ---
  k_zero<<<2, 256, 0, stream>>>(stats1, 512);
  k_zero<<<4, 256, 0, stream>>>(stats2, 1024);
  k_zero<<<32, 256, 0, stream>>>(pool, 8192);

  k_conv1<<<dim3(49, 256, 16), 256, 0, stream>>>(images, conv1w, conv1raw, stats1);
  k_bnfin<<<1, 256, 0, stream>>>(stats1, bn1g, bn1b, sc1, sh1, 256, invCnt);
  k_cvt<<<512, 256, 0, stream>>>(pww, pwW, 131072);
  k_dw<<<dim3(49, 256, 16), 256, 0, stream>>>(conv1raw, dww, sc1, sh1, dwA);

  // pointwise conv as WMMA GEMM: M=200704, N=512, K=256
  k_gemm_stats<<<dim3(8, 1568), 128, 0, stream>>>(dwA, 256, pwW, 256, 256, stats2, 512);
  k_bnfin<<<2, 256, 0, stream>>>(stats2, bn2g, bn2b, sc2, sh2, 512, invCnt);
  k_gemm_pool<<<dim3(8, 1568), 128, 0, stream>>>(dwA, pwW, 256, sc2, sh2, pool);
  k_scale<<<32, 256, 0, stream>>>(pool, 1.f / 12544.f, 8192);

  k_se<<<16, 128, 0, stream>>>(pool, se1, se2, encw, encb, vw, ctx);

  // --- decoder ---
  k_cvt<<<64000, 256, 0, stream>>>(fcw, fcW, 16384000);
  k_cvt<<<6144, 256, 0, stream>>>(wih, wihB, 1572864);
  k_gctx<<<dim3(6, 16), 256, 0, stream>>>(ctx, wih, bih, gctx);
  k_embed<<<512, 256, 0, stream>>>(caps, embed, aemb);
  // batched input projection: M=512 (b,t), N=1536, K=512 (emb columns of Wih)
  k_gemm_plain<<<dim3(24, 4), 128, 0, stream>>>(aemb, 512, wihB, 1024, 512,
                                                (const float*)nullptr, giemb, 1536);
  k_zero<<<32, 256, 0, stream>>>(hbuf, 8192);
  for (int t = 0; t < 32; t++)
    k_gru<<<16, 512, 0, stream>>>(giemb, gctx, whh, bhh, hbuf, hall, t);

  // logits: M=512 (b,t), N=32000, K=512, bf16 weights, +fc_b, fp32 out
  k_gemm_plain<<<dim3(500, 4), 128, 0, stream>>>(hall, 512, fcW, 512, 512,
                                                 fcb, out, 32000);
}
